// _DenseBlock_60335700574775
// MI455X (gfx1250) — compile-verified, matrix operand order checked by reading, not run on hardware
//
#include <hip/hip_runtime.h>
#include <hip/hip_fp16.h>

typedef __attribute__((ext_vector_type(16))) _Float16 v16h;
typedef __attribute__((ext_vector_type(8)))  float    v8f;
typedef __attribute__((ext_vector_type(4)))  unsigned u32x4;
typedef __attribute__((ext_vector_type(8)))  int      i32x8;
typedef __attribute__((ext_vector_type(4)))  int      i32x4;

#define B_        16
#define H_        56
#define W_        56
#define HW        3136
#define M_TOT     (B_*HW)     /* 50176 = 392*128 = 196*256 */
#define C_TOT     512
#define INTER     128
#define GROWTH    32
#define NLAYERS   12
#define QMAX_     255.0f
#define EPS_      1e-5f

#if __has_builtin(__builtin_amdgcn_tensor_load_to_lds) && __has_builtin(__builtin_amdgcn_s_wait_tensorcnt)
#define USE_TDM 1
#else
#define USE_TDM 0
#endif

/* ---------------- helpers ---------------- */

__device__ __forceinline__ float fq(float x, float mn, float mx) {
    float scale = fmaxf((mx - mn) * (1.0f / QMAX_), 1e-8f);
    float q = rintf((x - mn) / scale);
    q = fminf(fmaxf(q, 0.0f), QMAX_);
    return q * scale + mn;
}

#if USE_TDM
/* Issue a TDM 2D tile load: width x rows fp16 tile, row stride = stride
   (elements), global -> LDS (contiguous rows of `width` halves).
   D# layout per CDNA5 ISA ch.8 (groups 0/1; groups 2/3 zero => 2D).
   6-arg builtin form (clang-23 / therock-10.0 headers). */
__device__ __forceinline__ void tdm_load_tile(void* lds, const _Float16* gptr,
                                              int width, int rows, int stride) {
    /* generic AS pointer to shared memory: low 32 bits are the LDS offset */
    unsigned laddr = (unsigned)(unsigned long long)(size_t)lds;
    unsigned long long ga = (unsigned long long)(size_t)gptr;
    u32x4 g0 = { 1u,                               /* count = 1 valid descriptor  */
                 laddr,                            /* lds_addr [63:32]            */
                 (unsigned)(ga & 0xffffffffu),     /* global_addr [95:64]         */
                 (unsigned)((ga >> 32) & 0x1ffffffu) | 0x80000000u /* type=2 */ };
    i32x8 g1;
    g1[0] = 0x00010000;                            /* data_size=1 (2B), mask=0    */
    g1[1] = (int)((width & 0xffff) << 16);         /* tensor_dim0 [15:0]          */
    g1[2] = (int)(((unsigned)width >> 16) | ((unsigned)(rows & 0xffff) << 16));
    g1[3] = (int)(((unsigned)rows >> 16) | ((unsigned)(width & 0xffff) << 16)); /* tile_dim0 */
    g1[4] = rows & 0xffff;                         /* tile_dim1, tile_dim2=0      */
    g1[5] = stride;                                /* tensor_dim0_stride low 32   */
    g1[6] = 0;
    g1[7] = 0;
    i32x4 z4 = { 0, 0, 0, 0 };
    i32x8 z8 = { 0, 0, 0, 0, 0, 0, 0, 0 };
    __builtin_amdgcn_tensor_load_to_lds(g0, g1, z4, z4, z8, 0);
}
#endif

/* copy input x (NCHW, 128ch) into d_out channels [0,128) of the 512ch concat */
__global__ __launch_bounds__(256) void copy_in_k(const float* __restrict__ x,
                                                 float* __restrict__ out) {
    size_t total = (size_t)B_ * 128 * HW;
    for (size_t e = (size_t)blockIdx.x * 256 + threadIdx.x; e < total;
         e += (size_t)gridDim.x * 256) {
        size_t ni = e / (128 * HW);
        size_t r  = e % (128 * HW);
        out[(ni * C_TOT) * HW + r] = x[e];
    }
}

/* per-channel mean/var over first c channels of concat buffer */
__global__ __launch_bounds__(256) void chan_stats_k(const float* __restrict__ xin,
                                                    float* __restrict__ mean,
                                                    float* __restrict__ var) {
    __shared__ float s1[256], s2[256];
    const int ci = blockIdx.x;
    float sum = 0.f, sq = 0.f;
    for (int e = threadIdx.x; e < M_TOT; e += 256) {
        int ni = e / HW, p = e % HW;
        float v = xin[((size_t)ni * C_TOT + ci) * HW + p];
        sum += v; sq += v * v;
    }
    s1[threadIdx.x] = sum; s2[threadIdx.x] = sq; __syncthreads();
    for (int s = 128; s > 0; s >>= 1) {
        if (threadIdx.x < s) { s1[threadIdx.x] += s1[threadIdx.x + s];
                               s2[threadIdx.x] += s2[threadIdx.x + s]; }
        __syncthreads();
    }
    if (threadIdx.x == 0) {
        float mu = s1[0] / (float)M_TOT;
        mean[ci] = mu;
        var[ci]  = s2[0] / (float)M_TOT - mu * mu;
    }
}

__global__ void init_minmax_k(unsigned* mm) {
    mm[0] = 0x7F800000u;  /* +inf (post-ReLU values >=0 -> uint order works) */
    mm[1] = 0x00000000u;
}

__global__ __launch_bounds__(256) void bnrelu_minmax_k(const float* __restrict__ xin,
        const float* __restrict__ mean, const float* __restrict__ var,
        const float* __restrict__ g, const float* __restrict__ b,
        int c, unsigned* __restrict__ mm) {
    __shared__ float smn[256], smx[256];
    const int t = threadIdx.x;
    size_t total = (size_t)M_TOT * c;
    float lmn = 3.4e38f, lmx = 0.f;
    for (size_t e = (size_t)blockIdx.x * 256 + t; e < total;
         e += (size_t)gridDim.x * 256) {
        int ci = (int)(e % c);
        size_t m = e / c;
        int ni = (int)(m / HW), p = (int)(m % HW);
        float v = xin[((size_t)ni * C_TOT + ci) * HW + p];
        float y = (v - mean[ci]) * rsqrtf(var[ci] + EPS_) * g[ci] + b[ci];
        y = fmaxf(y, 0.f);
        lmn = fminf(lmn, y); lmx = fmaxf(lmx, y);
    }
    smn[t] = lmn; smx[t] = lmx; __syncthreads();
    for (int s = 128; s > 0; s >>= 1) {
        if (t < s) { smn[t] = fminf(smn[t], smn[t + s]);
                     smx[t] = fmaxf(smx[t], smx[t + s]); }
        __syncthreads();
    }
    if (t == 0) {
        atomicMin(&mm[0], __float_as_uint(smn[0]));
        atomicMax(&mm[1], __float_as_uint(smx[0]));
    }
}

__global__ __launch_bounds__(256) void bnrelu_quant_k(const float* __restrict__ xin,
        const float* __restrict__ mean, const float* __restrict__ var,
        const float* __restrict__ g, const float* __restrict__ b,
        int c, const unsigned* __restrict__ mm, _Float16* __restrict__ A1) {
    float mn = __uint_as_float(mm[0]);
    float mx = __uint_as_float(mm[1]);
    size_t total = (size_t)M_TOT * c;
    for (size_t e = (size_t)blockIdx.x * 256 + threadIdx.x; e < total;
         e += (size_t)gridDim.x * 256) {
        int ci = (int)(e % c);
        size_t m = e / c;
        int ni = (int)(m / HW), p = (int)(m % HW);
        float v = xin[((size_t)ni * C_TOT + ci) * HW + p];
        float y = (v - mean[ci]) * rsqrtf(var[ci] + EPS_) * g[ci] + b[ci];
        y = fmaxf(y, 0.f);
        A1[e] = (_Float16)fq(y, mn, mx);
    }
}

/* per-output-channel weight quant; w row-major [rows x K] -> fp16 same layout */
__global__ __launch_bounds__(256) void quant_w_rows_k(const float* __restrict__ w,
        int K, _Float16* __restrict__ outB) {
    __shared__ float smn[256], smx[256];
    const int co = blockIdx.x, t = threadIdx.x;
    float lmn = 3.4e38f, lmx = -3.4e38f;
    for (int k = t; k < K; k += 256) {
        float v = w[(size_t)co * K + k];
        lmn = fminf(lmn, v); lmx = fmaxf(lmx, v);
    }
    smn[t] = lmn; smx[t] = lmx; __syncthreads();
    for (int s = 128; s > 0; s >>= 1) {
        if (t < s) { smn[t] = fminf(smn[t], smn[t + s]);
                     smx[t] = fmaxf(smx[t], smx[t + s]); }
        __syncthreads();
    }
    float mn = smn[0], mx = smx[0];
    for (int k = t; k < K; k += 256)
        outB[(size_t)co * K + k] = (_Float16)fq(w[(size_t)co * K + k], mn, mx);
}

/* 3x3 weight quant + reorder: w [32][128][3][3] -> B2t [32][ rs*128 + ci ] */
__global__ __launch_bounds__(256) void quant_w2_k(const float* __restrict__ w,
                                                  _Float16* __restrict__ outB) {
    __shared__ float smn[256], smx[256];
    const int co = blockIdx.x, t = threadIdx.x;
    const int K = 128 * 9;
    float lmn = 3.4e38f, lmx = -3.4e38f;
    for (int k = t; k < K; k += 256) {
        float v = w[(size_t)co * K + k];
        lmn = fminf(lmn, v); lmx = fmaxf(lmx, v);
    }
    smn[t] = lmn; smx[t] = lmx; __syncthreads();
    for (int s = 128; s > 0; s >>= 1) {
        if (t < s) { smn[t] = fminf(smn[t], smn[t + s]);
                     smx[t] = fmaxf(smx[t], smx[t + s]); }
        __syncthreads();
    }
    float mn = smn[0], mx = smx[0];
    for (int k = t; k < K; k += 256) {
        int ci = k / 9, rs = k % 9;
        outB[(size_t)co * K + rs * 128 + ci] =
            (_Float16)fq(w[(size_t)co * K + k], mn, mx);
    }
}

/* --------- WMMA GEMM: C[M x 128] = A[M x K] * Bt[128 x K]^T (fp16 -> f32) ---------
   128x128 block tile, 8 waves in a 4(M)x2(N) grid, each wave 32x64.
   Tiles staged to double-buffered LDS by the Tensor Data Mover (wave 0 issues,
   s_wait_tensorcnt + barrier hands off). Falls back to register-staged copies. */
__global__ __launch_bounds__(256) void gemm_f16_n128_k(const _Float16* __restrict__ A,
        const _Float16* __restrict__ Bt, float* __restrict__ C, int K) {
    __shared__ _Float16 As[2][128 * 32];
    __shared__ _Float16 Bs[2][128 * 32];
    const int t = threadIdx.x;
    const int wave = t >> 5, lane = t & 31;
    const int wm = (wave >> 1) * 32;     /* wave's M offset in tile */
    const int wn = (wave & 1) * 64;      /* wave's N offset in tile */
    const int m0 = blockIdx.x * 128;
    const int frow = lane & 15;
    const int fseg = (lane >> 4) * 16;
    v8f acc[2][4] = {};

#if USE_TDM
    if (wave == 0) {
        tdm_load_tile(&As[0][0], A + (size_t)m0 * K, 32, 128, K);
        tdm_load_tile(&Bs[0][0], Bt, 32, 128, K);
    }
    int buf = 0;
    for (int k0 = 0; k0 < K; k0 += 32) {
        if (wave == 0) __builtin_amdgcn_s_wait_tensorcnt(0);
        __syncthreads();                     /* tile `buf` ready, prev reads done */
        if (wave == 0 && (k0 + 32) < K) {    /* prefetch next tile into buf^1 */
            tdm_load_tile(&As[buf ^ 1][0], A + (size_t)m0 * K + k0 + 32, 32, 128, K);
            tdm_load_tile(&Bs[buf ^ 1][0], Bt + k0 + 32, 32, 128, K);
        }
        v16h a0 = *(const v16h*)&As[buf][(wm + frow) * 32 + fseg];
        v16h a1 = *(const v16h*)&As[buf][(wm + 16 + frow) * 32 + fseg];
#pragma unroll
        for (int nt = 0; nt < 4; ++nt) {
            v16h b = *(const v16h*)&Bs[buf][(wn + nt * 16 + frow) * 32 + fseg];
            acc[0][nt] = __builtin_amdgcn_wmma_f32_16x16x32_f16(
                false, a0, false, b, (short)0, acc[0][nt], false, false);
            acc[1][nt] = __builtin_amdgcn_wmma_f32_16x16x32_f16(
                false, a1, false, b, (short)0, acc[1][nt], false, false);
        }
        buf ^= 1;
    }
#else
    const int ldRow = t >> 1;
    const int ldSeg = (t & 1) * 16;
    for (int k0 = 0; k0 < K; k0 += 32) {
        *(v16h*)&As[0][ldRow * 32 + ldSeg] =
            *(const v16h*)&A[(size_t)(m0 + ldRow) * K + k0 + ldSeg];
        *(v16h*)&Bs[0][ldRow * 32 + ldSeg] =
            *(const v16h*)&Bt[(size_t)ldRow * K + k0 + ldSeg];
        __syncthreads();
        v16h a0 = *(const v16h*)&As[0][(wm + frow) * 32 + fseg];
        v16h a1 = *(const v16h*)&As[0][(wm + 16 + frow) * 32 + fseg];
#pragma unroll
        for (int nt = 0; nt < 4; ++nt) {
            v16h b = *(const v16h*)&Bs[0][(wn + nt * 16 + frow) * 32 + fseg];
            acc[0][nt] = __builtin_amdgcn_wmma_f32_16x16x32_f16(
                false, a0, false, b, (short)0, acc[0][nt], false, false);
            acc[1][nt] = __builtin_amdgcn_wmma_f32_16x16x32_f16(
                false, a1, false, b, (short)0, acc[1][nt], false, false);
        }
        __syncthreads();
    }
#endif
    const int colLo = lane & 15;
    const int mAdd  = (lane >> 4) * 8;
#pragma unroll
    for (int mi = 0; mi < 2; ++mi)
#pragma unroll
        for (int nt = 0; nt < 4; ++nt)
#pragma unroll
            for (int r = 0; r < 8; ++r) {
                int mrow = m0 + wm + mi * 16 + mAdd + r;
                C[(size_t)mrow * 128 + wn + nt * 16 + colLo] = acc[mi][nt][r];
            }
}

/* per-column stats over C1 [M x 128] */
__global__ __launch_bounds__(256) void colstats_k(const float* __restrict__ C1,
        float* __restrict__ cmean, float* __restrict__ cmin, float* __restrict__ cmax) {
    __shared__ float s1[256], smn[256], smx[256];
    const int co = blockIdx.x, t = threadIdx.x;
    float sum = 0.f, lmn = 3.4e38f, lmx = -3.4e38f;
    for (int m = t; m < M_TOT; m += 256) {
        float v = C1[(size_t)m * 128 + co];
        sum += v; lmn = fminf(lmn, v); lmx = fmaxf(lmx, v);
    }
    s1[t] = sum; smn[t] = lmn; smx[t] = lmx; __syncthreads();
    for (int s = 128; s > 0; s >>= 1) {
        if (t < s) { s1[t] += s1[t + s];
                     smn[t] = fminf(smn[t], smn[t + s]);
                     smx[t] = fmaxf(smx[t], smx[t + s]); }
        __syncthreads();
    }
    if (t == 0) { cmean[co] = s1[0] / (float)M_TOT; cmin[co] = smn[0]; cmax[co] = smx[0]; }
}

__device__ __forceinline__ float q128(float v, float* sbuf) {
    const int t = threadIdx.x;
    sbuf[t] = v; __syncthreads();
    for (int s = 64; s > 0; s >>= 1) { if (t < s) sbuf[t] = fminf(sbuf[t], sbuf[t + s]); __syncthreads(); }
    float mn = sbuf[0]; __syncthreads();
    sbuf[t] = v; __syncthreads();
    for (int s = 64; s > 0; s >>= 1) { if (t < s) sbuf[t] = fmaxf(sbuf[t], sbuf[t + s]); __syncthreads(); }
    float mx = sbuf[0]; __syncthreads();
    return fq(v, mn, mx);
}

__global__ __launch_bounds__(128) void rbn_prep_k(const float* __restrict__ cmin,
        const float* __restrict__ cmax, const float* __restrict__ g2,
        const float* __restrict__ b2, float* __restrict__ qscale,
        float* __restrict__ qg, float* __restrict__ qb) {
    __shared__ float sbuf[128];
    const int t = threadIdx.x;
    const float scale_fix = 0.175f * (1.0f + sqrtf(3.14159265358979f * 1.38629436111989f))
                          / sqrtf(2.0f * logf((float)M_TOT));
    qscale[t] = q128((cmax[t] - cmin[t]) * scale_fix, sbuf);
    qg[t]     = q128(g2[t], sbuf);
    qb[t]     = q128(b2[t], sbuf);
}

__global__ __launch_bounds__(256) void rbn_minmax_k(const float* __restrict__ C1,
        const float* __restrict__ cmean, const float* __restrict__ qscale,
        const float* __restrict__ qg, const float* __restrict__ qb,
        unsigned* __restrict__ mm) {
    __shared__ float smn[256], smx[256];
    const int t = threadIdx.x;
    size_t total = (size_t)M_TOT * 128;
    float lmn = 3.4e38f, lmx = 0.f;
    for (size_t e = (size_t)blockIdx.x * 256 + t; e < total;
         e += (size_t)gridDim.x * 256) {
        int co = (int)(e & 127);
        float y = (C1[e] - cmean[co]) / (qscale[co] + EPS_) * qg[co] + qb[co];
        y = fmaxf(y, 0.f);
        lmn = fminf(lmn, y); lmx = fmaxf(lmx, y);
    }
    smn[t] = lmn; smx[t] = lmx; __syncthreads();
    for (int s = 128; s > 0; s >>= 1) {
        if (t < s) { smn[t] = fminf(smn[t], smn[t + s]);
                     smx[t] = fmaxf(smx[t], smx[t + s]); }
        __syncthreads();
    }
    if (t == 0) {
        atomicMin(&mm[0], __float_as_uint(smn[0]));
        atomicMax(&mm[1], __float_as_uint(smx[0]));
    }
}

__global__ __launch_bounds__(256) void rbn_quant_k(const float* __restrict__ C1,
        const float* __restrict__ cmean, const float* __restrict__ qscale,
        const float* __restrict__ qg, const float* __restrict__ qb,
        const unsigned* __restrict__ mm, _Float16* __restrict__ A2) {
    float mn = __uint_as_float(mm[0]);
    float mx = __uint_as_float(mm[1]);
    size_t total = (size_t)M_TOT * 128;
    for (size_t e = (size_t)blockIdx.x * 256 + threadIdx.x; e < total;
         e += (size_t)gridDim.x * 256) {
        int co = (int)(e & 127);
        float y = (C1[e] - cmean[co]) / (qscale[co] + EPS_) * qg[co] + qb[co];
        y = fmaxf(y, 0.f);
        A2[e] = (_Float16)fq(y, mn, mx);
    }
}

/* --------- 3x3 conv as implicit WMMA GEMM: A2 NHWC [M x 128], B2t [32][1152]
   256-row block tile, 8 waves each 32(M)x32(N); on-the-fly zero-padded im2col. */
__global__ __launch_bounds__(256) void conv3_wmma_k(const _Float16* __restrict__ A2,
        const _Float16* __restrict__ B2t, float* __restrict__ out, int cobase) {
    __shared__ _Float16 As[256 * 32];
    __shared__ _Float16 Bs[32 * 32];
    const int t = threadIdx.x, wave = t >> 5, lane = t & 31;
    const int wm = wave * 32;
    const int m0 = blockIdx.x * 256;
    const int frow = lane & 15;
    const int fseg = (lane >> 4) * 16;
    v8f acc[2][2] = {};
    const int m  = m0 + t;               /* each thread stages one row */
    const int ni = m / HW;
    const int p  = m % HW;
    const int y  = p / W_, x = p % W_;
    for (int rs = 0; rs < 9; ++rs) {
        const int yy = y + rs / 3 - 1;
        const int xx = x + rs % 3 - 1;
        const bool inb = (yy >= 0) && (yy < H_) && (xx >= 0) && (xx < W_);
        const _Float16* src = A2 + ((size_t)ni * HW + yy * W_ + xx) * 128;
        for (int kc = 0; kc < 4; ++kc) {
            v16h av0 = {}, av1 = {};
            if (inb) {
                av0 = *(const v16h*)&src[kc * 32];
                av1 = *(const v16h*)&src[kc * 32 + 16];
            }
            *(v16h*)&As[t * 32]      = av0;
            *(v16h*)&As[t * 32 + 16] = av1;
            if (t < 64) {
                int n = t >> 1, seg = (t & 1) * 16;
                *(v16h*)&Bs[n * 32 + seg] =
                    *(const v16h*)&B2t[(size_t)n * 1152 + rs * 128 + kc * 32 + seg];
            }
            __syncthreads();
            v16h a0 = *(const v16h*)&As[(wm + frow) * 32 + fseg];
            v16h a1 = *(const v16h*)&As[(wm + 16 + frow) * 32 + fseg];
#pragma unroll
            for (int nt = 0; nt < 2; ++nt) {
                v16h b = *(const v16h*)&Bs[(nt * 16 + frow) * 32 + fseg];
                acc[0][nt] = __builtin_amdgcn_wmma_f32_16x16x32_f16(
                    false, a0, false, b, (short)0, acc[0][nt], false, false);
                acc[1][nt] = __builtin_amdgcn_wmma_f32_16x16x32_f16(
                    false, a1, false, b, (short)0, acc[1][nt], false, false);
            }
            __syncthreads();
        }
    }
    const int colLo = lane & 15;
    const int mAdd  = (lane >> 4) * 8;
#pragma unroll
    for (int mi = 0; mi < 2; ++mi)
#pragma unroll
        for (int nt = 0; nt < 2; ++nt)
#pragma unroll
            for (int r = 0; r < 8; ++r) {
                int mrow = m0 + wm + mi * 16 + mAdd + r;
                int nim  = mrow / HW, pp = mrow % HW;
                int co   = cobase + nt * 16 + colLo;
                out[((size_t)nim * C_TOT + co) * HW + pp] = acc[mi][nt][r];
            }
}

/* ---------------- driver ---------------- */

extern "C" void kernel_launch(void* const* d_in, const int* in_sizes, int n_in,
                              void* d_out, int out_size, void* d_ws, size_t ws_size,
                              hipStream_t stream) {
    (void)in_sizes; (void)n_in; (void)out_size; (void)ws_size;
    const float* x = (const float*)d_in[0];
    float* out = (float*)d_out;
    char* ws = (char*)d_ws;

    /* workspace layout (bytes) */
    _Float16* A1   = (_Float16*)(ws + 0);                 /* <= 50176*512 fp16 */
    float*    C1   = (float*)   (ws + 51380224);          /* 50176*128 f32 */
    _Float16* A2   = (_Float16*)(ws + 77070336);          /* 50176*128 fp16 */
    _Float16* B1t  = (_Float16*)(ws + 89915392);          /* 128*512 fp16 */
    _Float16* B2t  = (_Float16*)(ws + 90046464);          /* 32*1152 fp16 */
    float*    mean = (float*)   (ws + 90120192);
    float*    var  = (float*)   (ws + 90122240);
    float*    cmean= (float*)   (ws + 90124288);
    float*    cmin = (float*)   (ws + 90124800);
    float*    cmax = (float*)   (ws + 90125312);
    float*    qscale=(float*)   (ws + 90125824);
    float*    qg   = (float*)   (ws + 90126336);
    float*    qb   = (float*)   (ws + 90126848);
    unsigned* mm   = (unsigned*)(ws + 90127360);

    const int EW_BLOCKS = 2048;
    const int MT1 = M_TOT / 128;   /* 392 tiles for the 1x1 GEMM  */
    const int MT3 = M_TOT / 256;   /* 196 tiles for the 3x3 conv  */

    copy_in_k<<<EW_BLOCKS, 256, 0, stream>>>(x, out);

    for (int i = 0; i < NLAYERS; ++i) {
        const int c = 128 + i * GROWTH;
        const float* g1 = (const float*)d_in[1 + 6 * i + 0];
        const float* b1 = (const float*)d_in[1 + 6 * i + 1];
        const float* w1 = (const float*)d_in[1 + 6 * i + 2];  /* [128][c] */
        const float* g2 = (const float*)d_in[1 + 6 * i + 3];
        const float* b2 = (const float*)d_in[1 + 6 * i + 4];
        const float* w2 = (const float*)d_in[1 + 6 * i + 5];  /* [32][128][3][3] */

        chan_stats_k<<<c, 256, 0, stream>>>(out, mean, var);
        init_minmax_k<<<1, 1, 0, stream>>>(mm);
        bnrelu_minmax_k<<<EW_BLOCKS, 256, 0, stream>>>(out, mean, var, g1, b1, c, mm);
        bnrelu_quant_k<<<EW_BLOCKS, 256, 0, stream>>>(out, mean, var, g1, b1, c, mm, A1);

        quant_w_rows_k<<<INTER, 256, 0, stream>>>(w1, c, B1t);
        gemm_f16_n128_k<<<MT1, 256, 0, stream>>>(A1, B1t, C1, c);

        colstats_k<<<INTER, 256, 0, stream>>>(C1, cmean, cmin, cmax);
        rbn_prep_k<<<1, 128, 0, stream>>>(cmin, cmax, g2, b2, qscale, qg, qb);
        init_minmax_k<<<1, 1, 0, stream>>>(mm);
        rbn_minmax_k<<<EW_BLOCKS, 256, 0, stream>>>(C1, cmean, qscale, qg, qb, mm);
        rbn_quant_k<<<EW_BLOCKS, 256, 0, stream>>>(C1, cmean, qscale, qg, qb, mm, A2);

        quant_w2_k<<<GROWTH, 256, 0, stream>>>(w2, B2t);
        conv3_wmma_k<<<MT3, 256, 0, stream>>>(A2, B2t, out, 128 + i * GROWTH);
    }
}
